// LRU_5944234737760
// MI455X (gfx1250) — matrix operand order repaired and checked by reading, not compiled, not run on hardware
//
#include <hip/hip_runtime.h>
#include <hip/hip_bf16.h>
#include <stdint.h>

// ---------------- problem dims ----------------
constexpr int Bdim = 8, Ldim = 8192, Hdim = 512, Ndim = 256;
constexpr int M     = Bdim * Ldim;   // 65536 rows (b,l)
constexpr int K     = Hdim;          // 512 (both GEMMs: H, and 2N)
constexpr int NOUT  = 2 * Ndim;      // 512 (re||im for GEMM1; H for GEMM2)
constexpr int CHUNK = 128;
constexpr int NCHUNK = Ldim / CHUNK; // 64
constexpr int KSTEPS = K / 32;       // 16

typedef __bf16 bf16_t;
typedef __bf16 v16bf __attribute__((ext_vector_type(16)));
typedef float  v8f   __attribute__((ext_vector_type(8)));

// ---------------- workspace layout (bytes) ----------------
constexpr size_t OFF_PARAM   = 0;                                   // 5*256 floats
constexpr size_t OFF_XBF     = 8192;
constexpr size_t SZ_XBF      = (size_t)M * Hdim * sizeof(bf16_t);   // 64 MiB
constexpr size_t OFF_W1      = OFF_XBF + SZ_XBF;
constexpr size_t SZ_W        = (size_t)NOUT * Hdim * sizeof(bf16_t);// 512 KiB
constexpr size_t OFF_W2      = OFF_W1 + SZ_W;
constexpr size_t OFF_BU      = OFF_W2 + SZ_W;
constexpr size_t SZ_BU       = (size_t)M * NOUT * sizeof(float);    // 128 MiB
constexpr size_t OFF_SBF     = OFF_BU + SZ_BU;
constexpr size_t SZ_SBF      = (size_t)M * NOUT * sizeof(bf16_t);   // 64 MiB
constexpr size_t OFF_CARRY   = OFF_SBF + SZ_SBF;
constexpr size_t SZ_CARRY    = (size_t)Bdim * NCHUNK * Ndim * sizeof(float2);
constexpr size_t OFF_CARRYIN = OFF_CARRY + SZ_CARRY;

// prm layout (floats): [0..255] lam_re, [256..511] lam_im,
// [512..767] lamP_re (=Re Λ^128), [768..1023] lamP_im, [1024..1279] gamma

// ---------------- parameter kernel ----------------
__global__ void lru_params(const float* __restrict__ nu_log,
                           const float* __restrict__ th_log,
                           float* __restrict__ prm) {
  int n = threadIdx.x;
  if (n < Ndim) {
    float nu  = __expf(nu_log[n]);
    float th  = __expf(th_log[n]);
    float r   = __expf(-nu);
    float lre = r * __cosf(th);
    float lim = r * __sinf(th);
    float gam = sqrtf(fmaxf(0.f, 1.f - r * r));
    float pre = lre, pim = lim;                  // Λ^(2^7) = Λ^128
#pragma unroll
    for (int i = 0; i < 7; ++i) {
      float t = pre * pre - pim * pim;
      pim = 2.f * pre * pim;
      pre = t;
    }
    prm[n]        = lre;
    prm[256 + n]  = lim;
    prm[512 + n]  = pre;
    prm[768 + n]  = pim;
    prm[1024 + n] = gam;
  }
}

// ---------------- x -> bf16 (vectorized) ----------------
__global__ void cvt_x(const float4* __restrict__ x, uint2* __restrict__ xb) {
  size_t i = (size_t)blockIdx.x * blockDim.x + threadIdx.x;
  float4 v = x[i];
  union { bf16_t h[4]; uint2 u; } p;
  p.h[0] = (bf16_t)v.x; p.h[1] = (bf16_t)v.y;
  p.h[2] = (bf16_t)v.z; p.h[3] = (bf16_t)v.w;
  xb[i] = p.u;
}

// ---------------- build bf16 weights ----------------
// W1[r][h], r<256: gamma[r]*B_re[r][h]; r>=256: gamma[r-256]*B_im[r-256][h]
__global__ void make_w1(const float* __restrict__ B_re, const float* __restrict__ B_im,
                        const float* __restrict__ prm, bf16_t* __restrict__ W1) {
  size_t i = (size_t)blockIdx.x * blockDim.x + threadIdx.x;  // < 512*512
  int r = (int)(i >> 9), h = (int)(i & 511);
  int n = r & 255;
  float g = prm[1024 + n];
  float v = (r < Ndim ? B_re[(size_t)n * Hdim + h] : B_im[(size_t)n * Hdim + h]) * g;
  W1[i] = (bf16_t)v;
}

// W2[h][k], k<256: C_re[h][k]; k>=256: -C_im[h][k-256]
__global__ void make_w2(const float* __restrict__ C_re, const float* __restrict__ C_im,
                        bf16_t* __restrict__ W2) {
  size_t i = (size_t)blockIdx.x * blockDim.x + threadIdx.x;
  int h = (int)(i >> 9), k = (int)(i & 511);
  float v = (k < Ndim) ? C_re[(size_t)h * Ndim + k] : -C_im[(size_t)h * Ndim + (k - Ndim)];
  W2[i] = (bf16_t)v;
}

// ---------------- bf16 WMMA GEMM: out[M,512] = A[M,512] @ Bw[512,512]^T ----------------
// MODE 0: plain fp32 store. MODE 1: out = acc + D[h]*Xres[m][h]  (final output)
// Double-buffered LDS: global->VGPR prefetch of next K-slice overlaps the WMMAs,
// one barrier per K-step.
template <int MODE>
__launch_bounds__(256)
__global__ void lru_gemm(const bf16_t* __restrict__ A, const bf16_t* __restrict__ Bw,
                         float* __restrict__ Cout,
                         const float* __restrict__ Xres, const float* __restrict__ Dvec) {
  __shared__ bf16_t As[2][128][40];   // 128x32 tile, +8 pad (80B rows keep 16B align)
  __shared__ bf16_t Bs[2][128][40];

  const int tid  = threadIdx.x;
  const int lane = tid & 31;
  const int wave = tid >> 5;
  const int wm   = wave & 1;   // wave row: 64 M-rows each
  const int wn   = wave >> 1;  // wave col: 32 N-cols each
  const int m0   = blockIdx.y * 128;
  const int n0   = blockIdx.x * 128;

  // staging: each thread owns 2 of the 512 16B chunks per tile, per matrix
  const int c0 = tid * 2, c1 = c0 + 1;
  const int r0 = c0 >> 2, q0 = c0 & 3;
  const int r1 = c1 >> 2, q1 = c1 & 3;
  const uint4* gA0 = reinterpret_cast<const uint4*>(A  + (size_t)(m0 + r0) * K) + q0;
  const uint4* gA1 = reinterpret_cast<const uint4*>(A  + (size_t)(m0 + r1) * K) + q1;
  const uint4* gB0 = reinterpret_cast<const uint4*>(Bw + (size_t)(n0 + r0) * K) + q0;
  const uint4* gB1 = reinterpret_cast<const uint4*>(Bw + (size_t)(n0 + r1) * K) + q1;
  // one K-step (32 bf16) = 4 uint4 along a row

  v8f acc[4][2];
  {
    v8f z = {};
#pragma unroll
    for (int i = 0; i < 4; ++i)
#pragma unroll
      for (int j = 0; j < 2; ++j) acc[i][j] = z;
  }

  const int mrow  = lane & 15;
  const int khalf = lane >> 4;

  // prologue: stage K-step 0 into buffer 0
  {
    uint4 pa0 = gA0[0], pa1 = gA1[0], pb0 = gB0[0], pb1 = gB1[0];
    *reinterpret_cast<uint4*>(&As[0][r0][q0 * 8]) = pa0;
    *reinterpret_cast<uint4*>(&As[0][r1][q1 * 8]) = pa1;
    *reinterpret_cast<uint4*>(&Bs[0][r0][q0 * 8]) = pb0;
    *reinterpret_cast<uint4*>(&Bs[0][r1][q1 * 8]) = pb1;
  }

  for (int ki = 0; ki < KSTEPS; ++ki) {
    __syncthreads();
    const int cur = ki & 1;

    uint4 ra0, ra1, rb0, rb1;
    if (ki + 1 < KSTEPS) {        // prefetch next K-slice while WMMAs run
      ra0 = gA0[(ki + 1) * 4];
      ra1 = gA1[(ki + 1) * 4];
      rb0 = gB0[(ki + 1) * 4];
      rb1 = gB1[(ki + 1) * 4];
    }

    // A fragment (16-bit A 16x32): lanes 0-15 -> K 0..7/16..23, lanes 16-31 -> K 8..15/24..31
    v16bf afr[4];
#pragma unroll
    for (int mf = 0; mf < 4; ++mf) {
      int r = wm * 64 + mf * 16 + mrow;
      union { uint4 u[2]; v16bf v; } fa;
      fa.u[0] = *reinterpret_cast<const uint4*>(&As[cur][r][khalf * 8]);
      fa.u[1] = *reinterpret_cast<const uint4*>(&As[cur][r][16 + khalf * 8]);
      afr[mf] = fa.v;
    }
    // B fragment (16-bit B 32x16): lanes 0-15 -> K 0..7/16..23, lanes 16-31 -> K 8..15/24..31, col = lane&15
    v16bf bfr[2];
#pragma unroll
    for (int nf = 0; nf < 2; ++nf) {
      int r = wn * 32 + nf * 16 + mrow;
      union { uint4 u[2]; v16bf v; } fb;
      fb.u[0] = *reinterpret_cast<const uint4*>(&Bs[cur][r][khalf * 8]);
      fb.u[1] = *reinterpret_cast<const uint4*>(&Bs[cur][r][16 + khalf * 8]);
      bfr[nf] = fb.v;
    }

#pragma unroll
    for (int mf = 0; mf < 4; ++mf)
#pragma unroll
      for (int nf = 0; nf < 2; ++nf)
        acc[mf][nf] = __builtin_amdgcn_wmma_f32_16x16x32_bf16(
            false, afr[mf], false, bfr[nf], (short)0, acc[mf][nf], false, false);

    if (ki + 1 < KSTEPS) {        // fill the other buffer; visible after next barrier
      const int nxt = cur ^ 1;
      *reinterpret_cast<uint4*>(&As[nxt][r0][q0 * 8]) = ra0;
      *reinterpret_cast<uint4*>(&As[nxt][r1][q1 * 8]) = ra1;
      *reinterpret_cast<uint4*>(&Bs[nxt][r0][q0 * 8]) = rb0;
      *reinterpret_cast<uint4*>(&Bs[nxt][r1][q1 * 8]) = rb1;
    }
  }

  // epilogue: 32-bit C/D layout — lane 0-15: M=r, lane 16-31: M=r+8; N = lane&15
  const int ncol   = lane & 15;
  const int rowoff = (lane >> 4) * 8;
#pragma unroll
  for (int mf = 0; mf < 4; ++mf) {
#pragma unroll
    for (int nf = 0; nf < 2; ++nf) {
      int h = n0 + wn * 32 + nf * 16 + ncol;
#pragma unroll
      for (int r = 0; r < 8; ++r) {
        int m = m0 + wm * 64 + mf * 16 + rowoff + r;
        float v = acc[mf][nf][r];
        if (MODE == 1) v += Dvec[h] * Xres[(size_t)m * Hdim + h];
        Cout[(size_t)m * NOUT + h] = v;
      }
    }
  }
}

// ---------------- chunked complex scan ----------------
// pass A: local scan within each 128-step chunk (in place on Bu), emit chunk-final state
__global__ void scan_local(float* __restrict__ Bu, float2* __restrict__ carry,
                           const float* __restrict__ prm) {
  int w    = blockIdx.x * 8 + (threadIdx.x >> 5);
  int lane = threadIdx.x & 31;
  int b     = w >> 9;        // 512 waves per batch
  int rem   = w & 511;
  int chunk = rem >> 3;
  int n     = (rem & 7) * 32 + lane;
  float lre = prm[n], lim = prm[256 + n];
  size_t row = ((size_t)b * Ldim + (size_t)chunk * CHUNK) * NOUT + n;
  float sre = 0.f, sim = 0.f;
#pragma unroll 4
  for (int t = 0; t < CHUNK; ++t) {
    __builtin_prefetch(&Bu[row + 8 * NOUT], 0, 0);         // hide HBM latency on the chain
    __builtin_prefetch(&Bu[row + 8 * NOUT + Ndim], 0, 0);
    float bre = Bu[row];
    float bim = Bu[row + Ndim];
    float nre = fmaf(lre, sre, fmaf(-lim, sim, bre));
    float nim = fmaf(lre, sim, fmaf( lim, sre, bim));
    sre = nre; sim = nim;
    Bu[row] = sre; Bu[row + Ndim] = sim;
    row += NOUT;
  }
  carry[((size_t)b * NCHUNK + chunk) * Ndim + n] = make_float2(sre, sim);
}

// pass B: exclusive scan of chunk carries: cin[c] = Λ^128 * cin[c-1] + fin[c-1]
__global__ void scan_carry(const float2* __restrict__ fin, float2* __restrict__ cin,
                           const float* __restrict__ prm) {
  int t = blockIdx.x * blockDim.x + threadIdx.x;   // 0..2047
  int b = t >> 8, n = t & 255;
  float pre = prm[512 + n], pim = prm[768 + n];
  float cre = 0.f, cim = 0.f;
  for (int c = 0; c < NCHUNK; ++c) {
    size_t idx = ((size_t)b * NCHUNK + c) * Ndim + n;
    cin[idx] = make_float2(cre, cim);
    float2 f = fin[idx];
    float nre = pre * cre - pim * cim + f.x;
    float nim = pre * cim + pim * cre + f.y;
    cre = nre; cim = nim;
  }
}

// pass C: state_t = local_t + Λ^{t+1} * carry_in ; emit bf16 states for GEMM2
__global__ void scan_fixup(const float* __restrict__ Bu, const float2* __restrict__ cin,
                           bf16_t* __restrict__ Sbf, const float* __restrict__ prm) {
  int w    = blockIdx.x * 8 + (threadIdx.x >> 5);
  int lane = threadIdx.x & 31;
  int b     = w >> 9;
  int rem   = w & 511;
  int chunk = rem >> 3;
  int n     = (rem & 7) * 32 + lane;
  float2 c  = cin[((size_t)b * NCHUNK + chunk) * Ndim + n];
  float lre = prm[n], lim = prm[256 + n];
  float pre = lre, pim = lim;  // running Λ^{t+1}
  size_t row = ((size_t)b * Ldim + (size_t)chunk * CHUNK) * NOUT + n;
#pragma unroll 4
  for (int t = 0; t < CHUNK; ++t) {
    __builtin_prefetch(&Bu[row + 8 * NOUT], 0, 0);
    __builtin_prefetch(&Bu[row + 8 * NOUT + Ndim], 0, 0);
    float sre = Bu[row]        + (pre * c.x - pim * c.y);
    float sim = Bu[row + Ndim] + (pre * c.y + pim * c.x);
    Sbf[row]        = (bf16_t)sre;
    Sbf[row + Ndim] = (bf16_t)sim;
    float npre = pre * lre - pim * lim;
    float npim = pre * lim + pim * lre;
    pre = npre; pim = npim;
    row += NOUT;
  }
}

// ---------------- launcher ----------------
extern "C" void kernel_launch(void* const* d_in, const int* in_sizes, int n_in,
                              void* d_out, int out_size, void* d_ws, size_t ws_size,
                              hipStream_t stream) {
  (void)in_sizes; (void)n_in; (void)out_size; (void)ws_size;
  const float* x      = (const float*)d_in[0];
  const float* nu_log = (const float*)d_in[1];
  const float* th_log = (const float*)d_in[2];
  const float* B_re   = (const float*)d_in[3];
  const float* B_im   = (const float*)d_in[4];
  const float* C_re   = (const float*)d_in[5];
  const float* C_im   = (const float*)d_in[6];
  const float* Dv     = (const float*)d_in[7];
  float* out = (float*)d_out;

  char*   ws    = (char*)d_ws;
  float*  prm   = (float*)(ws + OFF_PARAM);
  bf16_t* xbf   = (bf16_t*)(ws + OFF_XBF);
  bf16_t* W1    = (bf16_t*)(ws + OFF_W1);
  bf16_t* W2    = (bf16_t*)(ws + OFF_W2);
  float*  Bu    = (float*)(ws + OFF_BU);
  bf16_t* Sbf   = (bf16_t*)(ws + OFF_SBF);
  float2* carry = (float2*)(ws + OFF_CARRY);
  float2* cin   = (float2*)(ws + OFF_CARRYIN);

  lru_params<<<1, 256, 0, stream>>>(nu_log, th_log, prm);
  cvt_x<<<(M * Hdim / 4) / 256, 256, 0, stream>>>((const float4*)x, (uint2*)xbf);
  make_w1<<<(NOUT * Hdim) / 256, 256, 0, stream>>>(B_re, B_im, prm, W1);
  make_w2<<<(NOUT * Hdim) / 256, 256, 0, stream>>>(C_re, C_im, W2);

  // GEMM1: Bu = x_bf16 @ W1^T  (fp32 out)
  lru_gemm<0><<<dim3(NOUT / 128, M / 128), 256, 0, stream>>>(xbf, W1, Bu, nullptr, nullptr);

  // chunked scan over time
  scan_local<<<(Bdim * NCHUNK * 8) / 8, 256, 0, stream>>>(Bu, carry, prm);
  scan_carry<<<(Bdim * Ndim) / 256, 256, 0, stream>>>(carry, cin, prm);
  scan_fixup<<<(Bdim * NCHUNK * 8) / 8, 256, 0, stream>>>(Bu, cin, Sbf, prm);

  // GEMM2: out = states_bf16 @ W2^T + D*x
  lru_gemm<1><<<dim3(NOUT / 128, M / 128), 256, 0, stream>>>(Sbf, W2, out, x, Dv);
}
